// Block_33595234189829
// MI455X (gfx1250) — matrix operand order repaired and checked by reading, not compiled, not run on hardware
//
#include <hip/hip_runtime.h>
#include <hip/hip_bf16.h>

// ---------------------------------------------------------------------------
// Transformer block (pre-LN attn + FF), B=4 T=2048 E=1024 H=16 HS=64 FF=4096.
// bf16 WMMA (f32 accum) for all matmuls; async-to-LDS staged, double-buffered
// GEMM; f32 LN/softmax/epilogues.
// ---------------------------------------------------------------------------

#define Bsz 4
#define Tsz 2048
#define Esz 1024
#define Hn  16
#define HSz 64
#define FFsz 4096
#define BT (Bsz * Tsz)             // 8192 rows
#define QKV_STRIDE (3 * Hn * HSz)  // 3072

typedef unsigned short bf16raw;    // host-safe raw bf16 (kernel params)
typedef __attribute__((ext_vector_type(16))) __bf16 v16bf;
typedef __attribute__((ext_vector_type(8)))  float  v8f;

union F16Frag { uint4 q[2]; v16bf v; };

// Low 32 bits of a generic pointer into __shared__ = wave-relative LDS byte
// address (flat LDS aperture keeps the offset in addr[31:0]).
__device__ __forceinline__ unsigned lds_addr32(const void* p) {
    return (unsigned)(unsigned long long)p;
}

__device__ __forceinline__ void async_load_b128(unsigned lds, unsigned voff,
                                                const void* sbase) {
    // GLOBAL_LOAD_ASYNC_TO_LDS_B128: LDS[lds] <- MEM[sbase + voff] (16B/lane),
    // tracked by ASYNCcnt, no VGPR data movement.
    asm volatile("global_load_async_to_lds_b128 %0, %1, %2"
                 :: "v"(lds), "v"(voff), "s"(sbase)
                 : "memory");
}

__device__ __forceinline__ void wait_async0() {
    asm volatile("s_wait_asynccnt 0x0" ::: "memory");
}

// ---------------------------------------------------------------------------
// Weight transpose + f32 -> bf16 convert:  Wt[n*K + k] = (bf16) W[k*N + n]
// ---------------------------------------------------------------------------
__global__ void transpose_bf16_kernel(const float* __restrict__ W,
                                      bf16raw* __restrict__ Wt_,
                                      int K, int N) {
    __bf16* Wt = (__bf16*)Wt_;
    __shared__ float tile[16][17];
    const int n0 = blockIdx.x * 16;
    const int k0 = blockIdx.y * 16;
    tile[threadIdx.y][threadIdx.x] =
        W[(size_t)(k0 + threadIdx.y) * N + n0 + threadIdx.x];
    __syncthreads();
    Wt[(size_t)(n0 + threadIdx.y) * K + k0 + threadIdx.x] =
        (__bf16)tile[threadIdx.x][threadIdx.y];
}

// ---------------------------------------------------------------------------
// LayerNorm over E=1024, one 256-thread block per row, bf16 output
// ---------------------------------------------------------------------------
__global__ void ln_bf16_kernel(const float* __restrict__ x,
                               const float* __restrict__ g,
                               const float* __restrict__ be,
                               bf16raw* __restrict__ out_) {
    __bf16* out = (__bf16*)out_;
    const int row = blockIdx.x;
    const int tid = threadIdx.x;
    const float4 v = ((const float4*)(x + (size_t)row * Esz))[tid];
    __shared__ float red[256];
    red[tid] = v.x + v.y + v.z + v.w;
    __syncthreads();
    for (int o = 128; o > 0; o >>= 1) {
        if (tid < o) red[tid] += red[tid + o];
        __syncthreads();
    }
    const float mu = red[0] * (1.0f / Esz);
    __syncthreads();
    const float dx = v.x - mu, dy = v.y - mu, dz = v.z - mu, dw = v.w - mu;
    red[tid] = dx * dx + dy * dy + dz * dz + dw * dw;
    __syncthreads();
    for (int o = 128; o > 0; o >>= 1) {
        if (tid < o) red[tid] += red[tid + o];
        __syncthreads();
    }
    const float inv = rsqrtf(red[0] * (1.0f / Esz) + 1e-5f);
    const int c = tid * 4;
    __bf16* orow = out + (size_t)row * Esz + c;
    orow[0] = (__bf16)(dx * inv * g[c + 0] + be[c + 0]);
    orow[1] = (__bf16)(dy * inv * g[c + 1] + be[c + 1]);
    orow[2] = (__bf16)(dz * inv * g[c + 2] + be[c + 2]);
    orow[3] = (__bf16)(dw * inv * g[c + 3] + be[c + 3]);
}

// ---------------------------------------------------------------------------
// WMMA GEMM: C[M,N] = A[M,K] (bf16 row-major) x Bt[N,K]^T (bf16, K-major)
// 128x128 workgroup tile, 8 waves (4x2), wave tile 32x64 (2x4 of 16x16).
// BLOCK_K=64 (two WMMA k-steps), double-buffered LDS, staged with
// global_load_async_to_lds_b128 overlapping WMMA of the other buffer.
// Branchless staging: last iteration re-stages offset 0 into the dead buffer.
// Requires M%128==0, N%128==0, K%64==0 (true for all uses here).
// ---------------------------------------------------------------------------
#define GSTRIDE 72   // 64 + 8 bf16 pad (144B rows: 16B-aligned)
#define GBUF    (128 * GSTRIDE)

__global__ __launch_bounds__(256) void gemm_bf16_kernel(
    const bf16raw* __restrict__ A_, const bf16raw* __restrict__ Bt_,
    const float* __restrict__ bias, const float* __restrict__ res,
    float* __restrict__ outF, bf16raw* __restrict__ outB_,
    int M, int N, int K, int relu) {
    const __bf16* A  = (const __bf16*)A_;
    const __bf16* Bt = (const __bf16*)Bt_;
    __bf16* outB = (__bf16*)outB_;

    __shared__ __align__(16) __bf16 As[2 * GBUF];
    __shared__ __align__(16) __bf16 Bs[2 * GBUF];

    const int tid  = threadIdx.x;
    const int lane = tid & 31;
    const int wid  = tid >> 5;
    const int ll = lane & 15;       // column / row-in-tile selector
    const int lh = lane >> 4;       // k-half / m-half selector
    const int bn = blockIdx.x * 128;
    const int bm = blockIdx.y * 128;
    const int wm = (wid & 3) * 32;  // wave M offset
    const int wn = (wid >> 2) * 64; // wave N offset

    // Per-thread staging geometry: 1024 16B-vectors per 128x64 tile,
    // 4 per thread per matrix.  row = v>>3, chunk c8 = v&7.
    const int srow = tid >> 3;           // rows for i: srow + i*32
    const int sc8  = tid & 7;
    const unsigned asb = lds_addr32(&As[0]);
    const unsigned bsb = lds_addr32(&Bs[0]);
    unsigned voff[4], aoff[4];
#pragma unroll
    for (int i = 0; i < 4; ++i) {
        const int row = srow + i * 32;
        voff[i] = (unsigned)((row * K + sc8 * 8) * 2);       // global byte off
        aoff[i] = (unsigned)((row * GSTRIDE + sc8 * 8) * 2); // LDS byte off
    }
    const __bf16* gA = A  + (size_t)bm * K;
    const __bf16* gB = Bt + (size_t)bn * K;

    v8f acc[2][4];
#pragma unroll
    for (int i = 0; i < 2; ++i)
#pragma unroll
        for (int j = 0; j < 4; ++j) { v8f z = {}; acc[i][j] = z; }

    // Prologue: stage k-tile 0 into buffer 0
#pragma unroll
    for (int i = 0; i < 4; ++i) {
        async_load_b128(asb + aoff[i], voff[i], gA);
        async_load_b128(bsb + aoff[i], voff[i], gB);
    }

    int buf = 0;
    for (int kt = 0; kt < K; kt += 64) {
        wait_async0();
        __syncthreads();   // staging of `buf` complete for all waves

        // Stage next tile into the other buffer, overlapped with WMMA below.
        // Branchless: last iteration redundantly stages offset 0 into the
        // buffer nobody will read (S_ENDPGM's implicit wait-idle drains it).
        {
            const int ktn = (kt + 64 < K) ? (kt + 64) : 0;
            const __bf16* nA = gA + ktn;
            const __bf16* nB = gB + ktn;
            const unsigned lb = (unsigned)((buf ^ 1) * GBUF * 2);
#pragma unroll
            for (int i = 0; i < 4; ++i) {
                async_load_b128(asb + lb + aoff[i], voff[i], nA);
                async_load_b128(bsb + lb + aoff[i], voff[i], nB);
            }
        }

        const __bf16* ab = &As[buf * GBUF];
        const __bf16* bb = &Bs[buf * GBUF];
#pragma unroll
        for (int c = 0; c < 2; ++c) {
            F16Frag af[2], bfr[4];
#pragma unroll
            for (int sm = 0; sm < 2; ++sm) {
                // A 16x32: lane ll = row; k = c*32 + {lh*8..+7, lh*8+16..+23}
                const __bf16* ar =
                    ab + (wm + sm * 16 + ll) * GSTRIDE + c * 32 + lh * 8;
                af[sm].q[0] = *(const uint4*)ar;
                af[sm].q[1] = *(const uint4*)(ar + 16);
            }
#pragma unroll
            for (int sn = 0; sn < 4; ++sn) {
                // B 32x16: lane ll = col; k = c*32 + {lh*8..+7, lh*8+16..+23}
                const __bf16* br =
                    bb + (wn + sn * 16 + ll) * GSTRIDE + c * 32 + lh * 8;
                bfr[sn].q[0] = *(const uint4*)br;
                bfr[sn].q[1] = *(const uint4*)(br + 16);
            }
#pragma unroll
            for (int sm = 0; sm < 2; ++sm)
#pragma unroll
                for (int sn = 0; sn < 4; ++sn)
                    acc[sm][sn] = __builtin_amdgcn_wmma_f32_16x16x32_bf16(
                        false, af[sm].v, false, bfr[sn].v,
                        (short)0, acc[sm][sn], false, false);
        }
        buf ^= 1;
        // A wave's ds_loads complete (dscnt waits precede its last wmma)
        // before it reaches the next barrier -> safe to overwrite `buf` then.
    }

    // Epilogue: C layout = lane ll is column, VGPR r is row (lh*8 + r)
#pragma unroll
    for (int sm = 0; sm < 2; ++sm)
#pragma unroll
        for (int sn = 0; sn < 4; ++sn) {
            const int col = bn + wn + sn * 16 + ll;
            const float bcol = bias ? bias[col] : 0.0f;
#pragma unroll
            for (int r = 0; r < 8; ++r) {
                const int row = bm + wm + sm * 16 + lh * 8 + r;
                float v = acc[sm][sn][r] + bcol;
                if (relu) v = fmaxf(v, 0.0f);
                const size_t idx = (size_t)row * N + col;
                if (res)  v += res[idx];
                if (outF) outF[idx] = v;
                if (outB) outB[idx] = (__bf16)v;
            }
        }
}

// ---------------------------------------------------------------------------
// WMMA flash attention (causal). grid = (T/128, H, B), 256 threads.
// Each wave owns 16 query rows; KV processed in 64-row blocks.
//   S = Q K^T  : A=Q frag (regs, loaded once from global), B=K (LDS row-major)
//   O += P V   : A=P (via per-wave LDS round-trip), B=V^T (LDS, transposed)
// ---------------------------------------------------------------------------
#define KSTRIDE 72   // 64 + 8 bf16 pad (144B rows)

__global__ __launch_bounds__(256) void flash_attn_kernel(
    const bf16raw* __restrict__ qkv_, bf16raw* __restrict__ out_) {
    const __bf16* qkv = (const __bf16*)qkv_;
    __bf16* out = (__bf16*)out_;

    const int tid  = threadIdx.x;
    const int lane = tid & 31;
    const int wid  = tid >> 5;
    const int ll = lane & 15;
    const int lh = lane >> 4;
    const int q0 = blockIdx.x * 128;
    const int h  = blockIdx.y;
    const int b  = blockIdx.z;

    const __bf16* base = qkv + (size_t)b * Tsz * QKV_STRIDE;
    const __bf16* Qp = base + h * HSz;
    const __bf16* Kp = base + Hn * HSz + h * HSz;
    const __bf16* Vp = base + 2 * Hn * HSz + h * HSz;

    __shared__ __align__(16) __bf16 Klds[64 * KSTRIDE];  // [kv][hs]
    __shared__ __align__(16) __bf16 Vt[64 * KSTRIDE];    // [hs][kv]
    __shared__ __align__(16) __bf16 Plds[8][16 * KSTRIDE];

    // Q fragments: A operand, rows = q0 + wid*16 + ll, HS=64 -> 2 k-chunks
    F16Frag aq[2];
    {
        const __bf16* qrow = Qp + (size_t)(q0 + wid * 16 + ll) * QKV_STRIDE;
#pragma unroll
        for (int c = 0; c < 2; ++c) {
            aq[c].q[0] = *(const uint4*)(qrow + c * 32 + lh * 8);
            aq[c].q[1] = *(const uint4*)(qrow + c * 32 + lh * 8 + 16);
        }
    }

    v8f o[4];
#pragma unroll
    for (int i = 0; i < 4; ++i) { v8f z = {}; o[i] = z; }
    float m_i[8], l_i[8];
#pragma unroll
    for (int r = 0; r < 8; ++r) { m_i[r] = -3.0e38f; l_i[r] = 0.0f; }

    const int kv_end = q0 + 128;   // causal: only blocks touching q <= q0+127
    for (int kv0 = 0; kv0 < kv_end; kv0 += 64) {
        __syncthreads();
        // ---- stage K (row-major) and V (transposed) ----
#pragma unroll
        for (int i = 0; i < 2; ++i) {
            const int v   = tid + i * 256;   // 512 vectors of 8 bf16
            const int row = v >> 3;          // kv row 0..63
            const int c8  = v & 7;
            const uint4 kd =
                *(const uint4*)(Kp + (size_t)(kv0 + row) * QKV_STRIDE + c8 * 8);
            *(uint4*)&Klds[row * KSTRIDE + c8 * 8] = kd;
            const uint4 vd =
                *(const uint4*)(Vp + (size_t)(kv0 + row) * QKV_STRIDE + c8 * 8);
            const __bf16* ve = (const __bf16*)&vd;
#pragma unroll
            for (int j = 0; j < 8; ++j)
                Vt[(c8 * 8 + j) * KSTRIDE + row] = ve[j];
        }
        __syncthreads();

        // ---- S = (Q K^T) * 1/sqrt(HS) ----
        v8f s[4];
#pragma unroll
        for (int i = 0; i < 4; ++i) { v8f z = {}; s[i] = z; }
#pragma unroll
        for (int c = 0; c < 2; ++c)
#pragma unroll
            for (int sn = 0; sn < 4; ++sn) {
                F16Frag bk;  // lane ll = kv col; k = hs, {lh*8..+7, lh*8+16..+23}
                const __bf16* kr =
                    &Klds[(sn * 16 + ll) * KSTRIDE + c * 32 + lh * 8];
                bk.q[0] = *(const uint4*)kr;
                bk.q[1] = *(const uint4*)(kr + 16);
                s[sn] = __builtin_amdgcn_wmma_f32_16x16x32_bf16(
                    false, aq[c].v, false, bk.v, (short)0, s[sn], false, false);
            }

        const int qrow_base = q0 + wid * 16 + lh * 8;
        const bool need_mask = (kv0 + 63) > (q0 + wid * 16);
#pragma unroll
        for (int sn = 0; sn < 4; ++sn) {
            const int col = kv0 + sn * 16 + ll;
#pragma unroll
            for (int r = 0; r < 8; ++r) {
                float val = s[sn][r] * 0.125f;   // HS^-0.5
                if (need_mask && col > (qrow_base + r)) val = -3.0e38f;
                s[sn][r] = val;
            }
        }

        // ---- online softmax (rows live in 16-lane groups; xor-reduce) ----
#pragma unroll
        for (int r = 0; r < 8; ++r) {
            float mx = fmaxf(fmaxf(s[0][r], s[1][r]), fmaxf(s[2][r], s[3][r]));
            mx = fmaxf(mx, __shfl_xor(mx, 1));
            mx = fmaxf(mx, __shfl_xor(mx, 2));
            mx = fmaxf(mx, __shfl_xor(mx, 4));
            mx = fmaxf(mx, __shfl_xor(mx, 8));
            const float mn = fmaxf(m_i[r], mx);
            const float sc = __expf(m_i[r] - mn);
            m_i[r] = mn;
            float rowsum = 0.0f;
#pragma unroll
            for (int sn = 0; sn < 4; ++sn) {
                const float p = __expf(s[sn][r] - mn);
                s[sn][r] = p;
                rowsum += p;
            }
            rowsum += __shfl_xor(rowsum, 1);
            rowsum += __shfl_xor(rowsum, 2);
            rowsum += __shfl_xor(rowsum, 4);
            rowsum += __shfl_xor(rowsum, 8);
            l_i[r] = l_i[r] * sc + rowsum;
#pragma unroll
            for (int sn = 0; sn < 4; ++sn) o[sn][r] *= sc;
        }

        // ---- P: C-layout regs -> A-layout via per-wave LDS ----
        __bf16* pw = &Plds[wid][0];
#pragma unroll
        for (int sn = 0; sn < 4; ++sn)
#pragma unroll
            for (int r = 0; r < 8; ++r)
                pw[(lh * 8 + r) * KSTRIDE + sn * 16 + ll] = (__bf16)s[sn][r];
        // (same-wave LDS RAW: DS ops are in-order; compiler inserts dscnt wait)

        // ---- O += P V ----
#pragma unroll
        for (int c = 0; c < 2; ++c) {
            F16Frag ap;
            const __bf16* prow = pw + ll * KSTRIDE + c * 32;
            ap.q[0] = *(const uint4*)(prow + lh * 8);
            ap.q[1] = *(const uint4*)(prow + lh * 8 + 16);
#pragma unroll
            for (int sn = 0; sn < 4; ++sn) {
                F16Frag bv;  // lane ll = hs col; k = kv, {lh*8..+7, lh*8+16..+23} in Vt
                const __bf16* vr =
                    &Vt[(sn * 16 + ll) * KSTRIDE + c * 32 + lh * 8];
                bv.q[0] = *(const uint4*)vr;
                bv.q[1] = *(const uint4*)(vr + 16);
                o[sn] = __builtin_amdgcn_wmma_f32_16x16x32_bf16(
                    false, ap.v, false, bv.v, (short)0, o[sn], false, false);
            }
        }
    }

    // ---- epilogue: O / l, concat heads, bf16 out [B,T,H*HS] ----
    __bf16* obase = out + (size_t)b * Tsz * Esz + h * HSz;
#pragma unroll
    for (int r = 0; r < 8; ++r) {
        const float inv = 1.0f / l_i[r];
        const int q = q0 + wid * 16 + lh * 8 + r;
        __bf16* orow = obase + (size_t)q * Esz;
#pragma unroll
        for (int sn = 0; sn < 4; ++sn)
            orow[sn * 16 + ll] = (__bf16)(o[sn][r] * inv);
    }
}

// ---------------------------------------------------------------------------
// Host orchestration
// ---------------------------------------------------------------------------
extern "C" void kernel_launch(void* const* d_in, const int* in_sizes, int n_in,
                              void* d_out, int out_size, void* d_ws,
                              size_t ws_size, hipStream_t stream) {
    (void)in_sizes; (void)n_in; (void)out_size; (void)ws_size;
    const float* x           = (const float*)d_in[0];
    const float* qkv_w       = (const float*)d_in[1];
    const float* attn_proj_w = (const float*)d_in[2];
    const float* attn_proj_b = (const float*)d_in[3];
    const float* ln1_g = (const float*)d_in[4];
    const float* ln1_b = (const float*)d_in[5];
    const float* ln2_g = (const float*)d_in[6];
    const float* ln2_b = (const float*)d_in[7];
    const float* ff_w1 = (const float*)d_in[8];
    const float* ff_b1 = (const float*)d_in[9];
    const float* ff_w2 = (const float*)d_in[10];
    const float* ff_b2 = (const float*)d_in[11];
    float* out = (float*)d_out;

    char* ws = (char*)d_ws;
    size_t off = 0;
    auto alloc = [&](size_t bytes) -> void* {
        void* p = ws + off;
        off += (bytes + 255) & ~(size_t)255;
        return p;
    };
    bf16raw* qkv_wt  = (bf16raw*)alloc((size_t)3 * Esz * Esz * 2);  // [3072,1024]
    bf16raw* proj_wt = (bf16raw*)alloc((size_t)Esz * Esz * 2);      // [1024,1024]
    bf16raw* ff1_wt  = (bf16raw*)alloc((size_t)FFsz * Esz * 2);     // [4096,1024]
    bf16raw* ff2_wt  = (bf16raw*)alloc((size_t)Esz * FFsz * 2);     // [1024,4096]
    bf16raw* h1      = (bf16raw*)alloc((size_t)BT * Esz * 2);
    bf16raw* qkv     = (bf16raw*)alloc((size_t)BT * QKV_STRIDE * 2);
    bf16raw* att     = (bf16raw*)alloc((size_t)BT * Esz * 2);
    float*   x2      = (float*)  alloc((size_t)BT * Esz * 4);
    bf16raw* h2      = (bf16raw*)alloc((size_t)BT * Esz * 2);
    bf16raw* ffh     = (bf16raw*)alloc((size_t)BT * FFsz * 2);

    const dim3 t16(16, 16);
    // weight convert+transpose: Wt[N,K] <- W[K,N]
    transpose_bf16_kernel<<<dim3(3 * Esz / 16, Esz / 16), t16, 0, stream>>>(
        qkv_w, qkv_wt, Esz, 3 * Esz);
    transpose_bf16_kernel<<<dim3(Esz / 16, Esz / 16), t16, 0, stream>>>(
        attn_proj_w, proj_wt, Esz, Esz);
    transpose_bf16_kernel<<<dim3(FFsz / 16, Esz / 16), t16, 0, stream>>>(
        ff_w1, ff1_wt, Esz, FFsz);
    transpose_bf16_kernel<<<dim3(Esz / 16, FFsz / 16), t16, 0, stream>>>(
        ff_w2, ff2_wt, FFsz, Esz);

    // h1 = LN1(x)
    ln_bf16_kernel<<<BT, 256, 0, stream>>>(x, ln1_g, ln1_b, h1);

    // qkv = h1 @ qkv_w            [8192,3072]
    gemm_bf16_kernel<<<dim3(3 * Esz / 128, BT / 128), 256, 0, stream>>>(
        h1, qkv_wt, nullptr, nullptr, nullptr, qkv, BT, 3 * Esz, Esz, 0);

    // attention
    flash_attn_kernel<<<dim3(Tsz / 128, Hn, Bsz), 256, 0, stream>>>(qkv, att);

    // x2 = x + att @ proj_w + proj_b
    gemm_bf16_kernel<<<dim3(Esz / 128, BT / 128), 256, 0, stream>>>(
        att, proj_wt, attn_proj_b, x, x2, nullptr, BT, Esz, Esz, 0);

    // h2 = LN2(x2)
    ln_bf16_kernel<<<BT, 256, 0, stream>>>(x2, ln2_g, ln2_b, h2);

    // ffh = relu(h2 @ ff_w1 + ff_b1)   [8192,4096]
    gemm_bf16_kernel<<<dim3(FFsz / 128, BT / 128), 256, 0, stream>>>(
        h2, ff1_wt, ff_b1, nullptr, nullptr, ffh, BT, FFsz, Esz, 1);

    // out = x2 + ffh @ ff_w2 + ff_b2
    gemm_bf16_kernel<<<dim3(Esz / 128, BT / 128), 256, 0, stream>>>(
        ffh, ff2_wt, ff_b2, x2, out, nullptr, BT, Esz, FFsz, 0);
}
